// sae_lin_64226940944892
// MI455X (gfx1250) — matrix operand order repaired and checked by reading, not compiled, not run on hardware
//
#include <hip/hip_runtime.h>
#include <hip/hip_bf16.h>

typedef __attribute__((ext_vector_type(16))) _Float16 v16h;
typedef __attribute__((ext_vector_type(8)))  float    v8f;

// Problem dims (match reference setup_inputs)
constexpr int  Bz  = 256;                 // batch
constexpr int  Cc  = 2;
constexpr int  Tt  = 100;
constexpr int  Dd  = Cc * 32 * 32;        // 2048
constexpr int  Mm  = Tt * Bz;             // 25600 GEMM rows
constexpr long TBD = (long)Tt * Bz * Dd;  // 52,428,800 elements

// GEMM tiling
constexpr int BM = 128, BN = 128, BK = 32;
constexpr int LDT = BK + 8;               // padded LDS row stride in halfs (80B)

// ---------------------------------------------------------------------------
// CDNA5 async copy helpers (ASYNCcnt path; ISA ch10 / 15.18.3)
// GVS addressing: mem_addr = SGPR64(saddr) + VGPR_I32(voff bytes) [+ ioffset]
// ---------------------------------------------------------------------------
__device__ __forceinline__ void async_load_b128_gvs(const void* sbase,
                                                    unsigned voff_bytes,
                                                    unsigned lds_off) {
  asm volatile("global_load_async_to_lds_b128 %0, %1, %2"
               :: "v"(lds_off), "v"(voff_bytes), "s"(sbase) : "memory");
}
__device__ __forceinline__ void wait_async_le4() {
  asm volatile("s_wait_asynccnt 4" ::: "memory");
}
__device__ __forceinline__ void wait_async_0() {
  asm volatile("s_wait_asynccnt 0" ::: "memory");
}
__device__ __forceinline__ unsigned lds_u32(const void* p) {
  return (unsigned)(unsigned long long)p;   // low 32 bits = LDS byte address
}

// ---------------------------------------------------------------------------
// 1) Layout transform: x[B,C,T,H,W] f32 -> xt[(t*B+b), d] f16, d=c*1024+h*32+w
// ---------------------------------------------------------------------------
__global__ __launch_bounds__(256) void pack_x_kernel(const float* __restrict__ x,
                                                     _Float16* __restrict__ xt) {
  size_t idx = (size_t)blockIdx.x * blockDim.x + threadIdx.x;
  if (idx >= (size_t)TBD) return;
  int d  = (int)(idx % Dd);
  int m  = (int)(idx / Dd);
  int b  = m % Bz;
  int t  = m / Bz;
  int c  = d >> 10;          // d / (H*W)
  int hw = d & 1023;
  size_t src = (((size_t)(b * Cc + c) * Tt + t) << 10) + hw;
  xt[idx] = (_Float16)x[src];
}

__global__ __launch_bounds__(256) void pack_w_kernel(const float* __restrict__ w,
                                                     _Float16* __restrict__ wh, int n) {
  int i = blockIdx.x * blockDim.x + threadIdx.x;
  if (i < n) wh[i] = (_Float16)w[i];
}

// ---------------------------------------------------------------------------
// 2) Tiled WMMA GEMM:  C[m,n] = sum_k A[m,k] * Wt[n,k] + bias[n]  (opt sigmoid)
//    A row-major [M,K] f16, Wt row-major [N,K] f16, C row-major [M,N] f32.
//    256 thr = 8 wave32; waves 2x4 -> each wave owns 64x32 of the 128x128 tile.
//    Double-buffered LDS filled by global_load_async_to_lds_b128 (ASYNCcnt),
//    software-pipelined: tile k+1 streams in while tile k feeds the WMMAs.
// ---------------------------------------------------------------------------
template <bool SIG, int M, int N, int K>
__global__ __launch_bounds__(256) void gemm_wmma_kernel(
    const _Float16* __restrict__ A, const _Float16* __restrict__ Wt,
    const float* __restrict__ bias, float* __restrict__ Cout) {
  __shared__ __align__(16) _Float16 As[2][BM * LDT];
  __shared__ __align__(16) _Float16 Bs[2][BN * LDT];

  const int tid   = threadIdx.x;
  const int lane  = tid & 31;
  const int wave  = tid >> 5;
  const int wm    = wave >> 2;      // 0..1 : M direction
  const int wn    = wave & 3;       // 0..3 : N direction
  const int l15   = lane & 15;
  const int lhalf = lane >> 4;

  const int blockM = blockIdx.y * BM;
  const int blockN = blockIdx.x * BN;

  // Staging: 2 threads per tile row, 16 halfs (32B) each -> 2 async b128 per matrix.
  const int srow = tid >> 1;
  const int scol = (tid & 1) * 16;

  // Per-thread byte offsets (loop-invariant); scalar base advances by K-step.
  const unsigned voffA =
      (unsigned)(((size_t)(blockM + srow) * K + scol) * sizeof(_Float16));
  const unsigned voffB =
      (unsigned)(((size_t)(blockN + srow) * K + scol) * sizeof(_Float16));
  const unsigned laBase = lds_u32(&As[0][srow * LDT + scol]);
  const unsigned lbBase = lds_u32(&Bs[0][srow * LDT + scol]);
  constexpr unsigned bufStride = (unsigned)(BM * LDT * sizeof(_Float16));

  auto issue = [&](int kt, unsigned buf) {
    const _Float16* sa = A  + kt * BK;   // uniform -> SGPR pair
    const _Float16* sb = Wt + kt * BK;
    const unsigned la = laBase + buf * bufStride;
    const unsigned lb = lbBase + buf * bufStride;
    async_load_b128_gvs(sa, voffA,      la);
    async_load_b128_gvs(sa, voffA + 16, la + 16);
    async_load_b128_gvs(sb, voffB,      lb);
    async_load_b128_gvs(sb, voffB + 16, lb + 16);
  };

  v8f acc[4][2] = {};

  auto compute = [&](unsigned cur) {
    union Frag { uint4 q[2]; v16h v; };
    Frag af[4], bf[2];
    // B fragment (32x16, 16-bit): lanes 0-15 hold K 0..15, lanes 16-31 K 16..31
    #pragma unroll
    for (int nj = 0; nj < 2; ++nj) {
      const _Float16* p = &Bs[cur][(wn * 32 + nj * 16 + l15) * LDT + lhalf * 16];
      bf[nj].q[0] = *(const uint4*)p;
      bf[nj].q[1] = *(const uint4*)(p + 8);
    }
    // A fragment (16x32, 16-bit): lanes 0-15 hold K 0..15, lanes 16-31 K 16..31
    #pragma unroll
    for (int mi = 0; mi < 4; ++mi) {
      const _Float16* p = &As[cur][(wm * 64 + mi * 16 + l15) * LDT + lhalf * 16];
      af[mi].q[0] = *(const uint4*)p;
      af[mi].q[1] = *(const uint4*)(p + 8);
    }
    #pragma unroll
    for (int mi = 0; mi < 4; ++mi)
      #pragma unroll
      for (int nj = 0; nj < 2; ++nj)
        acc[mi][nj] = __builtin_amdgcn_wmma_f32_16x16x32_f16(
            false, af[mi].v, false, bf[nj].v, (short)0, acc[mi][nj], false, false);
  };

  constexpr int NK = K / BK;         // 64 (even)
  issue(0, 0);
  #pragma unroll 1
  for (int kt = 0; kt < NK - 2; kt += 2) {
    issue(kt + 1, 1); wait_async_le4();
    __syncthreads(); compute(0); __syncthreads();
    issue(kt + 2, 0); wait_async_le4();
    __syncthreads(); compute(1); __syncthreads();
  }
  issue(NK - 1, 1); wait_async_le4();
  __syncthreads(); compute(0);
  wait_async_0();
  __syncthreads(); compute(1);

  // Epilogue: C 16x16 f32 layout: col = lane&15, row = vgpr + 8*(lane>>4)
  #pragma unroll
  for (int mi = 0; mi < 4; ++mi) {
    #pragma unroll
    for (int nj = 0; nj < 2; ++nj) {
      int col = blockN + wn * 32 + nj * 16 + l15;
      float bv = bias[col];
      #pragma unroll
      for (int v = 0; v < 8; ++v) {
        int row = blockM + wm * 64 + mi * 16 + lhalf * 8 + v;
        float r = acc[mi][nj][v] + bv;
        if (SIG) r = 1.0f / (1.0f + __expf(-r));
        Cout[(size_t)row * N + col] = r;
      }
    }
  }
}

// ---------------------------------------------------------------------------
// 3) LIF scan: mem = beta*mem + e_t - spk_prev ; spk = (mem - 1 > 0)
//    Writes spk f32 to d_out (already in [T,B,D] order) and f16 copy for GEMM.
// ---------------------------------------------------------------------------
__global__ __launch_bounds__(256) void lif_scan_kernel(const float* __restrict__ enc,
                                                       const float* __restrict__ beta,
                                                       float* __restrict__ spk_f32,
                                                       _Float16* __restrict__ spk_f16) {
  int i = blockIdx.x * blockDim.x + threadIdx.x;  // 0 .. Bz*Dd
  int d = i & (Dd - 1);
  float bc = fminf(fmaxf(beta[d], 0.0f), 1.0f);
  float mem = 0.0f, sp = 0.0f;
  const size_t stride = (size_t)Bz * Dd;
  for (int t = 0; t < Tt; ++t) {
    float e = enc[(size_t)t * stride + i];
    mem = bc * mem + e - sp;
    sp = (mem - 1.0f) > 0.0f ? 1.0f : 0.0f;
    spk_f32[(size_t)t * stride + i] = sp;
    spk_f16[(size_t)t * stride + i] = (_Float16)sp;
  }
}

// ---------------------------------------------------------------------------
extern "C" void kernel_launch(void* const* d_in, const int* in_sizes, int n_in,
                              void* d_out, int out_size, void* d_ws, size_t ws_size,
                              hipStream_t stream) {
  const float* x     = (const float*)d_in[0];
  const float* W_enc = (const float*)d_in[1];
  const float* b_enc = (const float*)d_in[2];
  const float* W_dec = (const float*)d_in[3];
  const float* b_dec = (const float*)d_in[4];
  const float* beta  = (const float*)d_in[5];

  float* out     = (float*)d_out;
  float* spk_f32 = out;            // first T*B*D floats
  float* dec_f32 = out + TBD;      // second T*B*D floats

  // Workspace layout (~316 MiB):
  //   [0)       xt f16 (reused as spk f16 after encoder GEMM)
  //   [XT)      W_enc f16
  //   [XT+W)    W_dec f16
  //   [XT+2W)   enc f32
  char* ws = (char*)d_ws;
  const size_t XT_BYTES = (size_t)Mm * Dd * sizeof(_Float16);
  const size_t W_BYTES  = (size_t)Dd * Dd * sizeof(_Float16);
  _Float16* xt   = (_Float16*)ws;
  _Float16* wenc = (_Float16*)(ws + XT_BYTES);
  _Float16* wdec = (_Float16*)(ws + XT_BYTES + W_BYTES);
  float*    enc  = (float*)(ws + XT_BYTES + 2 * W_BYTES);

  // 1) pack inputs to f16
  {
    int blocks = (int)((TBD + 255) / 256);
    pack_x_kernel<<<blocks, 256, 0, stream>>>(x, xt);
  }
  {
    int n = Dd * Dd;
    int blocks = (n + 255) / 256;
    pack_w_kernel<<<blocks, 256, 0, stream>>>(W_enc, wenc, n);
    pack_w_kernel<<<blocks, 256, 0, stream>>>(W_dec, wdec, n);
  }

  dim3 gg(Dd / BN, Mm / BM);  // 16 x 200 workgroups

  // 2) encoder GEMM (f16 WMMA, f32 accum) -> enc
  gemm_wmma_kernel<false, Mm, Dd, Dd><<<gg, 256, 0, stream>>>(xt, wenc, b_enc, enc);

  // 3) LIF scan -> spk f32 (output) + spk f16 (reuse xt region)
  lif_scan_kernel<<<(Bz * Dd) / 256, 256, 0, stream>>>(enc, beta, spk_f32, xt);

  // 4) decoder GEMM + sigmoid -> dec output
  gemm_wmma_kernel<true, Mm, Dd, Dd><<<gg, 256, 0, stream>>>(xt, wdec, b_dec, dec_f32);
}
